// HybridConFormerSF_18571438588121
// MI455X (gfx1250) — compile-verified, matrix operand order checked by reading, not run on hardware
//
#include <hip/hip_runtime.h>

// ---------------- model constants ----------------
#define BB    2
#define TT    512
#define CCH   256     // DIM
#define DII   512     // 2*DIM inner mamba width
#define DSS   64
#define DTRN  16
#define NHEAD 4
#define WINW  128
#define PADW  63
#define LWIN  511     // T + 2*PAD - WIN + 1
#define TPAD  638     // T + 2*PAD
#define BT    (BB*TT)

typedef __bf16 bf16;
typedef __attribute__((ext_vector_type(16))) __bf16 v16bf;
typedef __attribute__((ext_vector_type(8)))  float  v8f;

#if defined(__has_builtin)
#if __has_builtin(__builtin_amdgcn_tensor_load_to_lds) && __has_builtin(__builtin_amdgcn_s_wait_tensorcnt)
#define HAVE_TDM 1
#endif
#endif
#ifndef HAVE_TDM
#define HAVE_TDM 0
#endif

// k index inside a 32-wide bf16 WMMA fragment for element e, lane-half grp.
// Per CDNA5 ISA 16-bit A layout: VGPR0..3 hold K pairs {0..7|8..15}, VGPR4..7 hold {16..23|24..31}.
__device__ __forceinline__ int kmap(int e, int grp) {
  int v = e >> 1, h = e & 1;
  int base = (v < 4) ? (2 * v + h) : (16 + 2 * (v - 4) + h);
  return base + grp * 8;
}

__device__ __forceinline__ v8f vzero8() {
  v8f z;
#pragma unroll
  for (int i = 0; i < 8; ++i) z[i] = 0.f;
  return z;
}

__device__ __forceinline__ float hannf(int j) {
  return 0.5f * (1.f - __cosf(6.28318530718f * (float)j / 127.f));
}

template <int ACT>
__device__ __forceinline__ float act_fn(float x) {
  if constexpr (ACT == 1) return 0.5f * x * (1.f + erff(x * 0.70710678118f)); // exact gelu
  else if constexpr (ACT == 2) return x / (1.f + __expf(-x));                // silu
  else if constexpr (ACT == 3) return fmaxf(x, 0.f);                         // relu
  else if constexpr (ACT == 4) return 1.f / (1.f + __expf(-x));              // sigmoid
  else if constexpr (ACT == 5) return (x > 20.f) ? x : log1pf(__expf(x));    // softplus
  else return x;
}

#if HAVE_TDM
typedef __attribute__((ext_vector_type(4))) unsigned int v4u;
typedef __attribute__((ext_vector_type(8))) int v8i;
typedef __attribute__((ext_vector_type(4))) int v4i;

// Low 32 bits of a generic pointer into LDS are the workgroup-relative LDS byte offset
// (flat->LDS mapping truncates to addr[31:0], ISA 10.2).
__device__ __forceinline__ unsigned ldsByteOff(const void* p) {
  return (unsigned)(size_t)p;
}

// Issue one TDM 2D tile load: tile_dim0=32 elems (bf16), tile_dim1=64 rows, row stride =
// strideElems. tensor_dim0/1 are set to the elements/rows remaining from the tile start so
// out-of-bounds reads are zero-filled by the TDM (handles ragged M/N/K edges for free).
__device__ __forceinline__ void tdm_load_tile(unsigned ldsOff, const bf16* gsrc,
                                              int remRows, int remCols, int strideElems) {
  unsigned long long ga = (unsigned long long)(size_t)(const void*)gsrc;
  unsigned td0 = (unsigned)remCols;
  unsigned td1 = (unsigned)remRows;
  v4u g0;
  g0[0] = 1u;                                                  // D# count = 1 (user mode)
  g0[1] = ldsOff;                                              // lds_addr (bytes)
  g0[2] = (unsigned)ga;                                        // global_addr[31:0]
  g0[3] = (unsigned)((ga >> 32) & 0x01FFFFFFu) | (2u << 30);   // global_addr[56:32] | type=2
  v8i g1;
  g1[0] = (int)(1u << 16);                                     // data_size = 1 (2 bytes)
  g1[1] = (int)((td0 & 0xFFFFu) << 16);                        // tensor_dim0[15:0]
  g1[2] = (int)((td0 >> 16) | ((td1 & 0xFFFFu) << 16));        // tensor_dim0[31:16] | tensor_dim1[15:0]
  g1[3] = (int)((td1 >> 16) | (32u << 16));                    // tensor_dim1[31:16] | tile_dim0=32
  g1[4] = 64;                                                  // tile_dim1=64, tile_dim2=0
  g1[5] = strideElems;                                         // tensor_dim0_stride[31:0]
  g1[6] = 0;                                                   // stride hi | dim1_stride lo
  g1[7] = 0;
  v4i gz;
  gz[0] = 0; gz[1] = 0; gz[2] = 0; gz[3] = 0;
#if defined(__clang_major__) && (__clang_major__ >= 23)
  v8i gz8;
#pragma unroll
  for (int i = 0; i < 8; ++i) gz8[i] = 0;
  __builtin_amdgcn_tensor_load_to_lds(g0, g1, gz, gz, gz8, 0);
#else
  __builtin_amdgcn_tensor_load_to_lds(g0, g1, gz, gz, 0);
#endif
}
#endif // HAVE_TDM

// ---------------- generic bf16 WMMA GEMM:  C[M,N] = act(A[M,K] * W[N,K]^T + bias) ----------------
// block = 256 threads = 8 waves; block tile 64x64; wave tile 16x32 (2 accumulators); BK = 32.
// Tile staging via Tensor Data Mover (double-buffered LDS, TENSORcnt) when available.
template <int ACT>
__global__ __launch_bounds__(256, 2)
void k_gemm(const bf16* __restrict__ A, const bf16* __restrict__ W,
            const float* __restrict__ bias, float* __restrict__ C,
            int M, int N, int K) {
  const int tid = threadIdx.x, lane = tid & 31, wave = tid >> 5;
  const int wm = (wave >> 1) * 16;   // 0,16,32,48
  const int wn = (wave & 1) * 32;    // 0,32
  const int grp = lane >> 4, l15 = lane & 15;
  const int bm = blockIdx.y * 64, bn = blockIdx.x * 64;
  v8f acc0 = vzero8(), acc1 = vzero8();
#if HAVE_TDM
  __shared__ bf16 sA[2][64 * 32];
  __shared__ bf16 sB[2][64 * 32];
  const bool issuer = (wave == 0);
  if (issuer) {
    tdm_load_tile(ldsByteOff(&sA[0][0]), A + (long long)bm * K, M - bm, K, K);
    tdm_load_tile(ldsByteOff(&sB[0][0]), W + (long long)bn * K, N - bn, K, K);
  }
  const int nk = (K + 31) >> 5;
  for (int ki = 0; ki < nk; ++ki) {
    const int cur = ki & 1;
    __builtin_amdgcn_s_wait_tensorcnt(0);   // tile `cur` resident (no-op for non-issuer waves)
    __syncthreads();
    if (issuer && (ki + 1 < nk)) {          // prefetch next tile into the other buffer
      int k0n = (ki + 1) * 32;
      tdm_load_tile(ldsByteOff(&sA[cur ^ 1][0]), A + (long long)bm * K + k0n, M - bm, K - k0n, K);
      tdm_load_tile(ldsByteOff(&sB[cur ^ 1][0]), W + (long long)bn * K + k0n, N - bn, K - k0n, K);
    }
    v16bf a, b0, b1;
#pragma unroll
    for (int e = 0; e < 16; ++e) {
      int kk = kmap(e, grp);
      a[e]  = sA[cur][(wm + l15) * 32 + kk];
      b0[e] = sB[cur][(wn + l15) * 32 + kk];
      b1[e] = sB[cur][(wn + 16 + l15) * 32 + kk];
    }
    acc0 = __builtin_amdgcn_wmma_f32_16x16x32_bf16(false, a, false, b0, (short)0, acc0, false, false);
    acc1 = __builtin_amdgcn_wmma_f32_16x16x32_bf16(false, a, false, b1, (short)0, acc1, false, false);
  }
#else
  __shared__ bf16 sA[64 * 32];
  __shared__ bf16 sB[64 * 32];
  const int r = tid >> 2, kb = (tid & 3) * 8;
  for (int k0 = 0; k0 < K; k0 += 32) {
#pragma unroll
    for (int i = 0; i < 8; ++i) {
      int gk = k0 + kb + i;
      bf16 av = (bf16)0.f, bv = (bf16)0.f;
      if (gk < K) {
        int gm = bm + r;
        if (gm < M) av = A[(long long)gm * K + gk];
        int gn = bn + r;
        if (gn < N) bv = W[(long long)gn * K + gk];
      }
      sA[r * 32 + kb + i] = av;
      sB[r * 32 + kb + i] = bv;
    }
    if (k0 + 32 < K) { // software prefetch of next k-tile -> global_prefetch_b8
      if (bm + r < M) __builtin_prefetch(&A[(long long)(bm + r) * K + k0 + 32 + kb], 0, 1);
      if (bn + r < N) __builtin_prefetch(&W[(long long)(bn + r) * K + k0 + 32 + kb], 0, 1);
    }
    __syncthreads();
    v16bf a, b0, b1;
#pragma unroll
    for (int e = 0; e < 16; ++e) {
      int kk = kmap(e, grp);
      a[e]  = sA[(wm + l15) * 32 + kk];
      b0[e] = sB[(wn + l15) * 32 + kk];
      b1[e] = sB[(wn + 16 + l15) * 32 + kk];
    }
    acc0 = __builtin_amdgcn_wmma_f32_16x16x32_bf16(false, a, false, b0, (short)0, acc0, false, false);
    acc1 = __builtin_amdgcn_wmma_f32_16x16x32_bf16(false, a, false, b1, (short)0, acc1, false, false);
    __syncthreads();
  }
#endif
#pragma unroll
  for (int v = 0; v < 8; ++v) {
    int row = bm + wm + v + 8 * grp;
    if (row < M) {
      int c0 = bn + wn + l15;
      if (c0 < N) {
        float x = acc0[v] + (bias ? bias[c0] : 0.f);
        C[(long long)row * N + c0] = act_fn<ACT>(x);
      }
      int c1 = c0 + 16;
      if (c1 < N) {
        float x = acc1[v] + (bias ? bias[c1] : 0.f);
        C[(long long)row * N + c1] = act_fn<ACT>(x);
      }
    }
  }
}

// ---------------- windowed attention (per window, per head), WMMA QK^T and PV ----------------
// G: [B][T][768] = LN(x) @ Wqkv^T (no bias; hann+bias folded in here).
// Output: overlap-added head outputs (pre out-projection) into Oacc [B][TPAD][256] via f32 atomics.
__global__ __launch_bounds__(256, 1)
void k_attn(const float* __restrict__ G, const float* __restrict__ qb,
            float* __restrict__ Oacc) {
  __shared__ __align__(16) char smem[65536];
  bf16* sK = (bf16*)smem;               // [128][64]
  bf16* sV = (bf16*)(smem + 16384);     // [128][64]
  bf16* sP = (bf16*)(smem + 32768);     // [128][128]
  const int w = blockIdx.x, head = blockIdx.y;
  const int b = w / LWIN, l = w % LWIN;
  const int tid = threadIdx.x, lane = tid & 31, wave = tid >> 5;
  const int grp = lane >> 4, l15 = lane & 15;

  // stage hann-scaled K,V (bf16) for this (window, head)
  for (int idx = tid; idx < 128 * 64; idx += 256) {
    int j = idx >> 6, d = idx & 63;
    int t = l + j - PADW;
    float h = hannf(j);
    float kv = 0.f, vv = 0.f;
    if (t >= 0 && t < TT) {
      const float* gp = G + ((long long)b * TT + t) * 768;
      kv = gp[256 + head * 64 + d];
      vv = gp[512 + head * 64 + d];
    }
    sK[idx] = (bf16)(h * kv + qb[256 + head * 64 + d]);
    sV[idx] = (bf16)(h * vv + qb[512 + head * 64 + d]);
  }
  __syncthreads();

  // Q fragments for this wave's 16-row strip (row = query slot)
  const int qrow = wave * 16 + l15;
  const int tq = l + qrow - PADW;
  const float hq = hannf(qrow);
  const float* gq = (tq >= 0 && tq < TT) ? (G + ((long long)b * TT + tq) * 768 + head * 64) : nullptr;
  v16bf aq0, aq1;
#pragma unroll
  for (int e = 0; e < 16; ++e) {
    int kq = kmap(e, grp);
    float q0 = gq ? gq[kq] : 0.f;
    float q1 = gq ? gq[32 + kq] : 0.f;
    aq0[e] = (bf16)(hq * q0 + qb[head * 64 + kq]);
    aq1[e] = (bf16)(hq * q1 + qb[head * 64 + 32 + kq]);
  }

  // S = Q K^T : 8 column tiles x (K=64 -> 2 bf16 WMMA steps)
  v8f s[8];
#pragma unroll
  for (int t = 0; t < 8; ++t) s[t] = vzero8();
#pragma unroll
  for (int t = 0; t < 8; ++t) {
    v16bf bk0, bk1;
#pragma unroll
    for (int e = 0; e < 16; ++e) {
      int kq = kmap(e, grp);
      bk0[e] = sK[(t * 16 + l15) * 64 + kq];
      bk1[e] = sK[(t * 16 + l15) * 64 + 32 + kq];
    }
    s[t] = __builtin_amdgcn_wmma_f32_16x16x32_bf16(false, aq0, false, bk0, (short)0, s[t], false, false);
    s[t] = __builtin_amdgcn_wmma_f32_16x16x32_bf16(false, aq1, false, bk1, (short)0, s[t], false, false);
  }

  // row softmax (scale 1/sqrt(64)) in registers: each row lives in one lane-half across 8 accum tiles
  const float scale = 0.125f;
#pragma unroll
  for (int v = 0; v < 8; ++v) {
    float m = -1e30f;
#pragma unroll
    for (int t = 0; t < 8; ++t) m = fmaxf(m, s[t][v]);
#pragma unroll
    for (int off = 1; off < 16; off <<= 1) m = fmaxf(m, __shfl_xor(m, off, 32));
    float sum = 0.f;
#pragma unroll
    for (int t = 0; t < 8; ++t) {
      float p = __expf((s[t][v] - m) * scale);
      s[t][v] = p;
      sum += p;
    }
#pragma unroll
    for (int off = 1; off < 16; off <<= 1) sum += __shfl_xor(sum, off, 32);
    float inv = 1.f / sum;
    int row = wave * 16 + v + 8 * grp;
#pragma unroll
    for (int t = 0; t < 8; ++t) sP[row * 128 + t * 16 + l15] = (bf16)(s[t][v] * inv);
  }
  __syncthreads();

  // O = P V : 4 column tiles x (K=128 -> 4 bf16 WMMA steps)
  v16bf ap[4];
#pragma unroll
  for (int ks = 0; ks < 4; ++ks)
#pragma unroll
    for (int e = 0; e < 16; ++e)
      ap[ks][e] = sP[(wave * 16 + l15) * 128 + ks * 32 + kmap(e, grp)];
  v8f o[4];
#pragma unroll
  for (int t = 0; t < 4; ++t) o[t] = vzero8();
#pragma unroll
  for (int ks = 0; ks < 4; ++ks) {
#pragma unroll
    for (int t = 0; t < 4; ++t) {
      v16bf bv;
#pragma unroll
      for (int e = 0; e < 16; ++e)
        bv[e] = sV[(ks * 32 + kmap(e, grp)) * 64 + t * 16 + l15];
      o[t] = __builtin_amdgcn_wmma_f32_16x16x32_bf16(false, ap[ks], false, bv, (short)0, o[t], false, false);
    }
  }
  // overlap-add into padded accumulator (out-projection deferred: it is linear)
#pragma unroll
  for (int t = 0; t < 4; ++t)
#pragma unroll
    for (int v = 0; v < 8; ++v) {
      int row = wave * 16 + v + 8 * grp;
      int p = l + row;
      int ch = head * 64 + t * 16 + l15;
      atomicAdd(&Oacc[((long long)b * TPAD + p) * CCH + ch], o[t][v]);
    }
}

// ---------------- elementwise / small kernels ----------------
__global__ void k_cvt(const float* __restrict__ s, bf16* __restrict__ d, long long n) {
  long long i = (long long)blockIdx.x * 256 + threadIdx.x;
  if (i < n) d[i] = (bf16)s[i];
}
__global__ void k_cvt2d(const float* __restrict__ s, int sld, int scol,
                        bf16* __restrict__ d, int dld, int dcol, int rows, int cols) {
  long long i = (long long)blockIdx.x * 256 + threadIdx.x;
  if (i >= (long long)rows * cols) return;
  int r = (int)(i / cols), c = (int)(i % cols);
  d[(long long)r * dld + dcol + c] = (bf16)s[(long long)r * sld + scol + c];
}
__global__ void k_fill(float* __restrict__ p, float v, long long n) {
  long long i = (long long)blockIdx.x * 256 + threadIdx.x;
  if (i < n) p[i] = v;
}
__global__ void k_comb(float* __restrict__ x, const float* __restrict__ y,
                       float a, float b, long long n) {
  long long i = (long long)blockIdx.x * 256 + threadIdx.x;
  if (i < n) x[i] = a * x[i] + b * y[i];
}
__global__ __launch_bounds__(256) void k_ln(const float* __restrict__ x, const float* __restrict__ g,
                                            const float* __restrict__ b, float* __restrict__ y) {
  int row = blockIdx.x, c = threadIdx.x;
  __shared__ float red[256];
  float v = x[(long long)row * CCH + c];
  red[c] = v;
  __syncthreads();
  for (int off = 128; off; off >>= 1) { if (c < off) red[c] += red[c + off]; __syncthreads(); }
  float mu = red[0] / (float)CCH;
  __syncthreads();
  float d = v - mu;
  red[c] = d * d;
  __syncthreads();
  for (int off = 128; off; off >>= 1) { if (c < off) red[c] += red[c + off]; __syncthreads(); }
  float var = red[0] / (float)CCH;
  y[(long long)row * CCH + c] = d * rsqrtf(var + 1e-5f) * g[c] + b[c];
}
// mamba causal depthwise conv (DC=4) over xc half of xz, + silu
__global__ void k_mamba_conv(const float* __restrict__ xz, const float* __restrict__ cw,
                             const float* __restrict__ cb, float* __restrict__ xc) {
  long long i = (long long)blockIdx.x * 256 + threadIdx.x;
  if (i >= (long long)BT * DII) return;
  int ch = (int)(i % DII);
  long long bt = i / DII;
  int t = (int)(bt % TT), b = (int)(bt / TT);
  float acc = cb[ch];
#pragma unroll
  for (int k = 0; k < 4; ++k) {
    int tt = t + k - 3;
    if (tt >= 0) acc += xz[((long long)b * TT + tt) * (2 * DII) + ch] * cw[ch * 4 + k];
  }
  xc[i] = acc / (1.f + __expf(-acc));
}
// selective scan: one block per (b, channel d), 64 threads = one SSM state each
__global__ __launch_bounds__(64) void k_scan(const float* __restrict__ dt, const float* __restrict__ xdb,
                                             const float* __restrict__ xc, const float* __restrict__ xz,
                                             const float* __restrict__ Alog, const float* __restrict__ Dp,
                                             float* __restrict__ ys) {
  int blk = blockIdx.x;
  int b = blk / DII, d = blk % DII;
  int s = threadIdx.x, wid = s >> 5, lane = s & 31;
  __shared__ float red[2];
  float Ad = -__expf(Alog[d * DSS + s]);
  float h = 0.f;
  for (int t = 0; t < TT; ++t) {
    long long r = (long long)b * TT + t;
    float dtv = dt[r * DII + d];
    float Bv = xdb[r * 144 + 16 + s];
    float Cv = xdb[r * 144 + 80 + s];
    float xv = xc[r * DII + d];
    h = __expf(dtv * Ad) * h + dtv * Bv * xv;
    float pr = h * Cv;
#pragma unroll
    for (int off = 16; off; off >>= 1) pr += __shfl_xor(pr, off, 32);
    if (lane == 0) red[wid] = pr;
    __syncthreads();
    if (s == 0) {
      float y = red[0] + red[1];
      float zz = xz[r * (2 * DII) + DII + d];
      ys[r * DII + d] = (y + Dp[d] * xv) * (zz / (1.f + __expf(-zz)));
    }
    __syncthreads();
  }
}
__global__ void k_flip(const float* __restrict__ x, float* __restrict__ y) {
  long long i = (long long)blockIdx.x * 256 + threadIdx.x;
  if (i >= (long long)BT * CCH) return;
  int c = (int)(i % CCH);
  long long bt = i / CCH;
  int t = (int)(bt % TT), b = (int)(bt / TT);
  y[i] = x[((long long)b * TT + (TT - 1 - t)) * CCH + c];
}
__global__ void k_roll(const float* __restrict__ x, float* __restrict__ y, int shift) {
  long long i = (long long)blockIdx.x * 256 + threadIdx.x;
  if (i >= (long long)BT * CCH) return;
  int c = (int)(i % CCH);
  long long bt = i / CCH;
  int t = (int)(bt % TT), b = (int)(bt / TT);
  int ts = (t - shift + TT) % TT;
  y[i] = x[((long long)b * TT + ts) * CCH + c];
}
__global__ void k_avg_roll(const float* __restrict__ ym, const float* __restrict__ ysr,
                           float* __restrict__ out, int shift) {
  long long i = (long long)blockIdx.x * 256 + threadIdx.x;
  if (i >= (long long)BT * CCH) return;
  int c = (int)(i % CCH);
  long long bt = i / CCH;
  int t = (int)(bt % TT), b = (int)(bt / TT);
  int ts = (t + shift) % TT;
  out[i] = 0.5f * (ym[i] + ysr[((long long)b * TT + ts) * CCH + c]);
}
// (Yproj over padded rows + counts*bias) / (counts + 1e-6), interior slice
__global__ void k_attn_norm(const float* __restrict__ Yp, const float* __restrict__ ob,
                            float* __restrict__ ym) {
  long long i = (long long)blockIdx.x * 256 + threadIdx.x;
  if (i >= (long long)BT * CCH) return;
  int c = (int)(i % CCH);
  long long bt = i / CCH;
  int t = (int)(bt % TT), b = (int)(bt / TT);
  int p = t + PADW;
  int lo = (p - WINW + 1 > 0) ? (p - WINW + 1) : 0;
  int hi = (p < LWIN - 1) ? p : (LWIN - 1);
  float cnt = (float)(hi - lo + 1);
  ym[i] = (Yp[((long long)b * TPAD + p) * CCH + c] + cnt * ob[c]) / (cnt + 1e-6f);
}
__global__ __launch_bounds__(256) void k_meanT(const float* __restrict__ x, float* __restrict__ s) {
  int b = blockIdx.x, c = threadIdx.x;
  float acc = 0.f;
  for (int t = 0; t < TT; ++t) acc += x[((long long)b * TT + t) * CCH + c];
  s[b * CCH + c] = acc / (float)TT;
}
__global__ void k_se_apply(float* __restrict__ x, const float* __restrict__ g) {
  long long i = (long long)blockIdx.x * 256 + threadIdx.x;
  if (i >= (long long)BT * CCH) return;
  int c = (int)(i % CCH);
  int b = (int)(i / ((long long)TT * CCH));
  x[i] *= g[b * CCH + c];
}
__global__ void k_se_apply_out(const float* __restrict__ x, const float* __restrict__ g,
                               float* __restrict__ out) {
  long long i = (long long)blockIdx.x * 256 + threadIdx.x;
  if (i >= (long long)BT * CCH) return;
  int c = (int)(i % CCH);
  int b = (int)(i / ((long long)TT * CCH));
  out[i] = x[i] * g[b * CCH + c];
}
__global__ void k_glu(const float* __restrict__ y, float* __restrict__ out) {
  long long i = (long long)blockIdx.x * 256 + threadIdx.x;
  if (i >= (long long)BT * CCH) return;
  int c = (int)(i % CCH);
  long long r = i / CCH;
  float a = y[r * 512 + c];
  float gg = y[r * 512 + 256 + c];
  out[i] = a / (1.f + __expf(-gg)); // a * sigmoid(g)
}
__global__ void k_dwconv(const float* __restrict__ y, const float* __restrict__ ww,
                         const float* __restrict__ wb, const float* __restrict__ bg,
                         const float* __restrict__ bb, float* __restrict__ out) {
  long long i = (long long)blockIdx.x * 256 + threadIdx.x;
  if (i >= (long long)BT * CCH) return;
  int c = (int)(i % CCH);
  long long bt = i / CCH;
  int t = (int)(bt % TT), b = (int)(bt / TT);
  float acc = wb[c];
#pragma unroll
  for (int k = 0; k < 3; ++k) {
    int tt = t + k - 1;
    if (tt >= 0 && tt < TT) acc += y[((long long)b * TT + tt) * CCH + c] * ww[c * 3 + k];
  }
  acc = acc * bg[c] * rsqrtf(1.f + 1e-5f) + bb[c];
  out[i] = acc / (1.f + __expf(-acc));
}

// ---------------- input index map (setup_inputs dict order) ----------------
enum {
  I_X = 0, I_BM_NG, I_BM_NB,
  I_MF_IN_W, I_MF_CONV_W, I_MF_CONV_B, I_MF_XPROJ_W, I_MF_DT_W, I_MF_DT_B, I_MF_ALOG, I_MF_D, I_MF_OUT_W,
  I_MB_IN_W, I_MB_CONV_W, I_MB_CONV_B, I_MB_XPROJ_W, I_MB_DT_W, I_MB_DT_B, I_MB_ALOG, I_MB_D, I_MB_OUT_W,
  I_BM_OUT_W, I_BM_OUT_B,
  I_F1_NG, I_F1_NB, I_F1_FC1_W, I_F1_FC1_B, I_F1_FC2_W, I_F1_FC2_B,
  I_F2_NG, I_F2_NB, I_F2_FC1_W, I_F2_FC1_B, I_F2_FC2_W, I_F2_FC2_B,
  I_MH_NG, I_MH_NB, I_MH_QKV_W, I_MH_QKV_B, I_MH_OUT_W, I_MH_OUT_B, I_MH_PROJ_W, I_MH_PROJ_B,
  I_MH_SE1_W, I_MH_SE2_W,
  I_CV_NG, I_CV_NB, I_CV_PW1_W, I_CV_PW1_B, I_CV_DW_W, I_CV_DW_B, I_CV_BN_G, I_CV_BN_B,
  I_CV_PW2_W, I_CV_PW2_B,
  I_BLK_SE1_W, I_BLK_SE2_W
};

extern "C" void kernel_launch(void* const* d_in, const int* in_sizes, int n_in,
                              void* d_out, int out_size, void* d_ws, size_t ws_size,
                              hipStream_t stream) {
  (void)in_sizes; (void)n_in; (void)out_size; (void)ws_size;
  char* wsp = (char*)d_ws;
  auto alloc = [&](size_t bytes) -> void* {
    void* p = (void*)wsp;
    wsp += (bytes + 255) & ~(size_t)255;
    return p;
  };
  auto F = [&](int i) -> const float* { return (const float*)d_in[i]; };
  auto nblk = [](long long n) { return (unsigned)((n + 255) / 256); };
  auto cvt = [&](const float* s, bf16* d, long long n) {
    k_cvt<<<nblk(n), 256, 0, stream>>>(s, d, n);
  };
  auto toBf = [&](int idx, long long n) -> bf16* {
    bf16* d = (bf16*)alloc(n * sizeof(bf16));
    cvt(F(idx), d, n);
    return d;
  };
  auto gemm = [&](int act, const bf16* A, const bf16* W, const float* bias, float* C,
                  int M, int N, int K) {
    dim3 g((N + 63) / 64, (M + 63) / 64);
    switch (act) {
      case 1: k_gemm<1><<<g, 256, 0, stream>>>(A, W, bias, C, M, N, K); break;
      case 2: k_gemm<2><<<g, 256, 0, stream>>>(A, W, bias, C, M, N, K); break;
      case 3: k_gemm<3><<<g, 256, 0, stream>>>(A, W, bias, C, M, N, K); break;
      case 4: k_gemm<4><<<g, 256, 0, stream>>>(A, W, bias, C, M, N, K); break;
      case 5: k_gemm<5><<<g, 256, 0, stream>>>(A, W, bias, C, M, N, K); break;
      default: k_gemm<0><<<g, 256, 0, stream>>>(A, W, bias, C, M, N, K); break;
    }
  };

  // ---- bf16 weight copies ----
  bf16* wInF   = toBf(I_MF_IN_W,   (long long)2 * DII * CCH);
  bf16* wXpF   = toBf(I_MF_XPROJ_W, 144LL * DII);
  bf16* wDtF   = toBf(I_MF_DT_W,   (long long)DII * DTRN);
  bf16* wOutF  = toBf(I_MF_OUT_W,  (long long)CCH * DII);
  bf16* wInB   = toBf(I_MB_IN_W,   (long long)2 * DII * CCH);
  bf16* wXpB   = toBf(I_MB_XPROJ_W, 144LL * DII);
  bf16* wDtB   = toBf(I_MB_DT_W,   (long long)DII * DTRN);
  bf16* wOutB  = toBf(I_MB_OUT_W,  (long long)CCH * DII);
  bf16* wBmOut = toBf(I_BM_OUT_W,  (long long)CCH * 2 * CCH);
  bf16* wFc1a  = toBf(I_F1_FC1_W,  4LL * CCH * CCH);
  bf16* wFc2a  = toBf(I_F1_FC2_W,  4LL * CCH * CCH);
  bf16* wFc1b  = toBf(I_F2_FC1_W,  4LL * CCH * CCH);
  bf16* wFc2b  = toBf(I_F2_FC2_W,  4LL * CCH * CCH);
  bf16* wQkv   = toBf(I_MH_QKV_W,  3LL * CCH * CCH);
  bf16* wMhOut = toBf(I_MH_OUT_W,  (long long)CCH * CCH);
  bf16* wProj  = toBf(I_MH_PROJ_W, (long long)CCH * CCH);
  bf16* wSe1m  = toBf(I_MH_SE1_W,  16LL * CCH);
  bf16* wSe2m  = toBf(I_MH_SE2_W,  (long long)CCH * 16);
  bf16* wPw1   = toBf(I_CV_PW1_W,  2LL * CCH * CCH);
  bf16* wPw2   = toBf(I_CV_PW2_W,  (long long)CCH * CCH);
  bf16* wSe1b  = toBf(I_BLK_SE1_W, 16LL * CCH);
  bf16* wSe2b  = toBf(I_BLK_SE2_W, (long long)CCH * 16);

  // ---- activation buffers ----
  float* xcur = (float*)alloc((size_t)BT * CCH * 4);
  float* xn   = (float*)alloc((size_t)BT * CCH * 4);
  float* xnf  = (float*)alloc((size_t)BT * CCH * 4);
  float* tln  = (float*)alloc((size_t)BT * CCH * 4);
  bf16*  bfc  = (bf16*)alloc((size_t)BT * CCH * 2);
  float* xz   = (float*)alloc((size_t)BT * 2 * DII * 4);
  float* xc   = (float*)alloc((size_t)BT * DII * 4);
  bf16*  bfdi = (bf16*)alloc((size_t)BT * DII * 2);
  float* xdb  = (float*)alloc((size_t)BT * 144 * 4);
  bf16*  bfdt = (bf16*)alloc((size_t)BT * DTRN * 2);
  float* dtb  = (float*)alloc((size_t)BT * DII * 4);
  float* ysc  = (float*)alloc((size_t)BT * DII * 4);
  float* xf   = (float*)alloc((size_t)BT * CCH * 4);
  float* xb   = (float*)alloc((size_t)BT * CCH * 4);
  float* xbt  = (float*)alloc((size_t)BT * CCH * 4);
  bf16*  bfcat = (bf16*)alloc((size_t)BT * 2 * CCH * 2);
  float* tmpc = (float*)alloc((size_t)BT * CCH * 4);
  float* Gq   = (float*)alloc((size_t)BT * 3 * CCH * 4);
  float* Oacc = (float*)alloc((size_t)BB * TPAD * CCH * 4);
  bf16*  bfo  = (bf16*)alloc((size_t)BB * TPAD * CCH * 2);
  float* Yp   = (float*)alloc((size_t)BB * TPAD * CCH * 4);
  float* ym   = (float*)alloc((size_t)BT * CCH * 4);
  float* ysr  = (float*)alloc((size_t)BT * CCH * 4);
  float* yavg = (float*)alloc((size_t)BT * CCH * 4);
  float* xs   = (float*)alloc((size_t)BT * CCH * 4);
  float* big  = (float*)alloc((size_t)BT * 4 * CCH * 4);
  bf16*  bfbig = (bf16*)alloc((size_t)BT * 4 * CCH * 2);
  float* tglu = (float*)alloc((size_t)BT * CCH * 4);
  float* tdw  = (float*)alloc((size_t)BT * CCH * 4);
  float* svec = (float*)alloc((size_t)BB * CCH * 4);
  bf16*  sbf  = (bf16*)alloc((size_t)BB * CCH * 2);
  float* smid = (float*)alloc((size_t)BB * 16 * 4);
  bf16*  smbf = (bf16*)alloc((size_t)BB * 16 * 2);
  float* gse  = (float*)alloc((size_t)BB * CCH * 4);

  const long long nBTC = (long long)BT * CCH;
  hipMemcpyAsync(xcur, d_in[I_X], (size_t)nBTC * 4, hipMemcpyDeviceToDevice, stream);

  // ---- mamba pipeline (shared for both directions) ----
  auto run_mamba = [&](const float* xin, bf16* wIn, bf16* wXp, bf16* wDt, bf16* wOut,
                       int icw, int icb, int idtb, int ia, int id, float* xout) {
    cvt(xin, bfc, nBTC);
    gemm(0, bfc, wIn, nullptr, xz, BT, 2 * DII, CCH);
    k_mamba_conv<<<nblk((long long)BT * DII), 256, 0, stream>>>(xz, F(icw), F(icb), xc);
    cvt(xc, bfdi, (long long)BT * DII);
    gemm(0, bfdi, wXp, nullptr, xdb, BT, 144, DII);
    k_cvt2d<<<nblk((long long)BT * DTRN), 256, 0, stream>>>(xdb, 144, 0, bfdt, DTRN, 0, BT, DTRN);
    gemm(5, bfdt, wDt, F(idtb), dtb, BT, DII, DTRN); // softplus(dt @ W^T + b)
    k_scan<<<BB * DII, 64, 0, stream>>>(dtb, xdb, xc, xz, F(ia), F(id), ysc);
    cvt(ysc, bfdi, (long long)BT * DII);
    gemm(0, bfdi, wOut, nullptr, xout, BT, CCH, DII);
  };

  // ---- windowed attention pass ----
  auto attend = [&](const float* xin, float* yout) {
    k_ln<<<BT, 256, 0, stream>>>(xin, F(I_MH_NG), F(I_MH_NB), tln);
    cvt(tln, bfc, nBTC);
    gemm(0, bfc, wQkv, nullptr, Gq, BT, 3 * CCH, CCH); // bias folded later (hann-aware)
    k_fill<<<nblk((long long)BB * TPAD * CCH), 256, 0, stream>>>(Oacc, 0.f, (long long)BB * TPAD * CCH);
    dim3 ga(BB * LWIN, NHEAD);
    k_attn<<<ga, 256, 0, stream>>>(Gq, F(I_MH_QKV_B), Oacc);
    cvt(Oacc, bfo, (long long)BB * TPAD * CCH);
    gemm(0, bfo, wMhOut, nullptr, Yp, BB * TPAD, CCH, CCH); // out-proj after overlap-add (linear)
    k_attn_norm<<<nblk(nBTC), 256, 0, stream>>>(Yp, F(I_MH_OUT_B), yout);
  };

  auto se_chain = [&](float* act, bf16* w1, bf16* w2) {
    k_meanT<<<BB, 256, 0, stream>>>(act, svec);
    cvt(svec, sbf, (long long)BB * CCH);
    gemm(3, sbf, w1, nullptr, smid, BB, 16, CCH);    // relu
    cvt(smid, smbf, (long long)BB * 16);
    gemm(4, smbf, w2, nullptr, gse, BB, CCH, 16);    // sigmoid
  };

  // ================= forward =================
  // 1) bidirectional mamba block
  k_ln<<<BT, 256, 0, stream>>>(xcur, F(I_BM_NG), F(I_BM_NB), xn);
  run_mamba(xn, wInF, wXpF, wDtF, wOutF, I_MF_CONV_W, I_MF_CONV_B, I_MF_DT_B, I_MF_ALOG, I_MF_D, xf);
  k_flip<<<nblk(nBTC), 256, 0, stream>>>(xn, xnf);
  run_mamba(xnf, wInB, wXpB, wDtB, wOutB, I_MB_CONV_W, I_MB_CONV_B, I_MB_DT_B, I_MB_ALOG, I_MB_D, xbt);
  k_flip<<<nblk(nBTC), 256, 0, stream>>>(xbt, xb);
  k_cvt2d<<<nblk(nBTC), 256, 0, stream>>>(xf, CCH, 0, bfcat, 2 * CCH, 0, BT, CCH);
  k_cvt2d<<<nblk(nBTC), 256, 0, stream>>>(xb, CCH, 0, bfcat, 2 * CCH, CCH, BT, CCH);
  gemm(0, bfcat, wBmOut, F(I_BM_OUT_B), tmpc, BT, CCH, 2 * CCH);
  k_comb<<<nblk(nBTC), 256, 0, stream>>>(xcur, tmpc, 1.f, 1.f, nBTC);

  // 2) FFN #1:  x = x + (x + 0.5*ffn(ln(x)))  ->  x = 2x + 0.5*y
  k_ln<<<BT, 256, 0, stream>>>(xcur, F(I_F1_NG), F(I_F1_NB), tln);
  cvt(tln, bfc, nBTC);
  gemm(1, bfc, wFc1a, F(I_F1_FC1_B), big, BT, 4 * CCH, CCH);   // gelu
  cvt(big, bfbig, (long long)BT * 4 * CCH);
  gemm(0, bfbig, wFc2a, F(I_F1_FC2_B), tmpc, BT, CCH, 4 * CCH);
  k_comb<<<nblk(nBTC), 256, 0, stream>>>(xcur, tmpc, 2.f, 0.5f, nBTC);

  // 3) MHSA (two shifted window passes)
  attend(xcur, ym);
  k_roll<<<nblk(nBTC), 256, 0, stream>>>(xcur, xs, WINW / 2);
  attend(xs, ysr);
  k_avg_roll<<<nblk(nBTC), 256, 0, stream>>>(ym, ysr, yavg, WINW / 2);
  cvt(yavg, bfc, nBTC);
  gemm(0, bfc, wProj, F(I_MH_PROJ_B), tmpc, BT, CCH, CCH);
  se_chain(tmpc, wSe1m, wSe2m);
  k_se_apply<<<nblk(nBTC), 256, 0, stream>>>(tmpc, gse);
  k_comb<<<nblk(nBTC), 256, 0, stream>>>(xcur, tmpc, 2.f, 1.f, nBTC);

  // 4) conv module:  x = 2x + conv(ln(x))
  k_ln<<<BT, 256, 0, stream>>>(xcur, F(I_CV_NG), F(I_CV_NB), tln);
  cvt(tln, bfc, nBTC);
  gemm(0, bfc, wPw1, F(I_CV_PW1_B), big, BT, 2 * CCH, CCH);
  k_glu<<<nblk(nBTC), 256, 0, stream>>>(big, tglu);
  k_dwconv<<<nblk(nBTC), 256, 0, stream>>>(tglu, F(I_CV_DW_W), F(I_CV_DW_B),
                                           F(I_CV_BN_G), F(I_CV_BN_B), tdw);
  cvt(tdw, bfc, nBTC);
  gemm(0, bfc, wPw2, F(I_CV_PW2_B), tmpc, BT, CCH, CCH);
  k_comb<<<nblk(nBTC), 256, 0, stream>>>(xcur, tmpc, 2.f, 1.f, nBTC);

  // 5) FFN #2
  k_ln<<<BT, 256, 0, stream>>>(xcur, F(I_F2_NG), F(I_F2_NB), tln);
  cvt(tln, bfc, nBTC);
  gemm(1, bfc, wFc1b, F(I_F2_FC1_B), big, BT, 4 * CCH, CCH);
  cvt(big, bfbig, (long long)BT * 4 * CCH);
  gemm(0, bfbig, wFc2b, F(I_F2_FC2_B), tmpc, BT, CCH, 4 * CCH);
  k_comb<<<nblk(nBTC), 256, 0, stream>>>(xcur, tmpc, 2.f, 0.5f, nBTC);

  // 6) final SE -> d_out
  se_chain(xcur, wSe1b, wSe2b);
  k_se_apply_out<<<nblk(nBTC), 256, 0, stream>>>(xcur, gse, (float*)d_out);
}
